// MusicAttrRegVAE_14963666059399
// MI455X (gfx1250) — matrix operand order repaired and checked by reading, not compiled, not run on hardware
//
#include <hip/hip_runtime.h>
#include <math.h>

// ---------------- constants ----------------
#define B_    32
#define T_    256
#define ROLL_ 342
#define RHY_  3
#define NOTE_ 16
#define CHR_  24
#define H_    512
#define Z_    128
#define G3H   1536   // 3*H
#define KX    352    // ROLL padded to mult of 32
#define KD    160    // (Z+RHY)=131 / (Z+NOTE)=144 padded
#define ZC    280    // 2Z + CHR
#define GIN   622    // ROLL + 2Z + CHR (cell_g input dim)

typedef __attribute__((ext_vector_type(16))) _Float16 v16h;
typedef __attribute__((ext_vector_type(8)))  _Float16 v8h;
typedef __attribute__((ext_vector_type(8)))  float    v8f;

// ---------------- WMMA helpers ----------------
// A fragment (16x32 f16): lane<16 -> row=lane, K = {0..7,16..23}; lane>=16 -> row=lane-16, K={8..15,24..31}
__device__ __forceinline__ v16h ldfragA(const _Float16* base, int ld) {
  int lane = threadIdx.x & 31;
  int row  = lane & 15;
  int kb   = (lane >> 4) << 3;            // 0 or 8
  const v8h lo = *(const v8h*)(base + (long)row * ld + kb);
  const v8h hi = *(const v8h*)(base + (long)row * ld + kb + 16);
  v16h r;
#pragma unroll
  for (int i = 0; i < 8; ++i) { r[i] = lo[i]; r[8 + i] = hi[i]; }
  return r;
}

// B fragment (32x16 f16): lane<16 -> col=lane, K={0..7,16..23}; lane>=16 -> col=lane-16, K={8..15,24..31}.
// Weight row n (row-major [N,K]) is B column n, contiguous along K.
__device__ __forceinline__ v16h ldfragB(const _Float16* base, int ld) {
  int lane = threadIdx.x & 31;
  int col  = lane & 15;
  int kb   = (lane >> 4) << 3;            // 0 or 8
  const v8h lo = *(const v8h*)(base + (long)col * ld + kb);
  const v8h hi = *(const v8h*)(base + (long)col * ld + kb + 16);
  v16h r;
#pragma unroll
  for (int i = 0; i < 8; ++i) { r[i] = lo[i]; r[8 + i] = hi[i]; }
  return r;
}

__device__ __forceinline__ v8f wmma32(v16h a, v16h b, v8f c) {
  return __builtin_amdgcn_wmma_f32_16x16x32_f16(false, a, false, b, (short)0, c, false, false);
}

__device__ __forceinline__ float sigmf(float x) { return 1.f / (1.f + __expf(-x)); }

// ---------------- staging kernels ----------------
__global__ void k_zero_h(_Float16* p, long n) {
  long i = (long)blockIdx.x * blockDim.x + threadIdx.x;
  if (i < n) p[i] = (_Float16)0.f;
}

__global__ void k_pack(const float* __restrict__ src, int sld, int soff,
                       _Float16* __restrict__ dst, int dld, int doff,
                       int rows, int cols) {
  long i = (long)blockIdx.x * blockDim.x + threadIdx.x;
  if (i >= (long)rows * cols) return;
  int r = (int)(i / cols), c = (int)(i % cols);
  dst[(long)r * dld + doff + c] = (_Float16)src[(long)r * sld + soff + c];
}

// build f16 decoder input [B*T, KD]: [seq | z | zero-pad]
__global__ void k_dec_in(const float* __restrict__ seq, int seqD,
                         const float* __restrict__ z, _Float16* __restrict__ dst) {
  long i = (long)blockIdx.x * blockDim.x + threadIdx.x;
  long total = (long)B_ * T_ * KD;
  if (i >= total) return;
  int  col = (int)(i % KD);
  long row = i / KD;
  int  b   = (int)(row / T_);
  float v = 0.f;
  if (col < seqD)            v = seq[row * seqD + col];
  else if (col < seqD + Z_)  v = z[b * Z_ + (col - seqD)];
  dst[i] = (_Float16)v;
}

// ---------------- generic WMMA GEMM: C[M,N] = A[M,Kp](f16) * W[N,Kp](f16)^T + bias ----------------
__global__ void __launch_bounds__(256)
k_gemm_f16(const _Float16* __restrict__ A, int lda,
           const _Float16* __restrict__ Bw, int ldb,
           const float* __restrict__ bias,
           float* __restrict__ C, int ldc, int M, int N, int Kp) {
  int wave = blockIdx.x * (blockDim.x >> 5) + (threadIdx.x >> 5);
  int ntn  = N >> 4;
  int mtiles = (M + 15) >> 4;
  if (wave >= mtiles * ntn) return;
  int mt = wave / ntn, nt = wave % ntn;
  const _Float16* Ab = A + (long)mt * 16 * lda;
  const _Float16* Bb = Bw + (long)nt * 16 * ldb;
  v8f acc = {};
  for (int k = 0; k < Kp; k += 32)
    acc = wmma32(ldfragA(Ab + k, lda), ldfragB(Bb + k, ldb), acc);
  int lane = threadIdx.x & 31;
  int col  = nt * 16 + (lane & 15);
  int rh   = (lane >> 4) * 8;
  float bv = bias ? bias[col] : 0.f;
#pragma unroll
  for (int i = 0; i < 8; ++i) {
    int row = mt * 16 + rh + i;
    C[(long)row * ldc + col] = acc[i] + bv;
  }
}

// ---------------- plain linear (small N / f32 path) ----------------
__global__ void k_linear(const float* __restrict__ A, int lda,
                         const float* __restrict__ W, int ldw, int wcoff,
                         const float* __restrict__ bias,
                         float* __restrict__ C, int ldc, int M, int N, int K) {
  long idx = (long)blockIdx.x * blockDim.x + threadIdx.x;
  if (idx >= (long)M * N) return;
  int m = (int)(idx / N), n = (int)(idx % N);
  const float* a = A + (long)m * lda;
  const float* w = W + (long)n * ldw + wcoff;
  float s = bias ? bias[n] : 0.f;
  for (int k = 0; k < K; ++k) s += a[k] * w[k];
  C[(long)m * ldc + n] = s;
}

// ---------------- persistent GRU over a sequence (one workgroup = 32 waves) ----------------
// gi: [B,T,3H] precomputed x@Wih^T+bih.  h kept in LDS (f16). Wave w owns H-columns [16w,16w+16).
__global__ void __launch_bounds__(1024)
k_gru_seq(const float* __restrict__ gi,
          const _Float16* __restrict__ Whh,   // [1536,512] f16
          const float* __restrict__ bhh,
          const float* __restrict__ h0,       // [B,H] or nullptr
          float* __restrict__ hs,             // [B,T,H] or nullptr
          float* __restrict__ hlast, int hl_ld, int hl_off,
          int Ts, int reverse) {
  __shared__ alignas(32) _Float16 hb[B_][H_];
  int tid = threadIdx.x;
  for (int i = tid; i < B_ * H_; i += 1024) {
    int b = i >> 9, c = i & (H_ - 1);
    hb[b][c] = (_Float16)(h0 ? h0[i] : 0.f);
  }
  __syncthreads();
  int w = tid >> 5, lane = tid & 31;
  int colL = lane & 15, rh = (lane >> 4) * 8;
  int c = w * 16 + colL;
  float bR = bhh[c], bZ = bhh[H_ + c], bN = bhh[2 * H_ + c];
  const _Float16* BR = Whh + (long)(w * 16) * H_;
  const _Float16* BZ = Whh + (long)(H_ + w * 16) * H_;
  const _Float16* BN = Whh + (long)(2 * H_ + w * 16) * H_;
  __builtin_prefetch(BR, 0, 1);
  __builtin_prefetch(BZ, 0, 1);
  __builtin_prefetch(BN, 0, 1);
  for (int t = 0; t < Ts; ++t) {
    int tt = reverse ? (Ts - 1 - t) : t;
    v8f aR0 = {}, aZ0 = {}, aN0 = {}, aR1 = {}, aZ1 = {}, aN1 = {};
    for (int k = 0; k < H_; k += 32) {
      v16h a0 = ldfragA(&hb[0][0] + k, H_);
      v16h a1 = ldfragA(&hb[16][0] + k, H_);
      v16h fR = ldfragB(BR + k, H_), fZ = ldfragB(BZ + k, H_), fN = ldfragB(BN + k, H_);
      aR0 = wmma32(a0, fR, aR0); aR1 = wmma32(a1, fR, aR1);
      aZ0 = wmma32(a0, fZ, aZ0); aZ1 = wmma32(a1, fZ, aZ1);
      aN0 = wmma32(a0, fN, aN0); aN1 = wmma32(a1, fN, aN1);
    }
    float hnew[2][8];
#pragma unroll
    for (int m = 0; m < 2; ++m) {
      v8f aR = m ? aR1 : aR0, aZ = m ? aZ1 : aZ0, aN = m ? aN1 : aN0;
#pragma unroll
      for (int i = 0; i < 8; ++i) {
        int b = m * 16 + rh + i;
        const float* g = gi + ((long)b * Ts + tt) * G3H;
        float r  = sigmf(g[c] + aR[i] + bR);
        float zg = sigmf(g[H_ + c] + aZ[i] + bZ);
        float n  = tanhf(g[2 * H_ + c] + r * (aN[i] + bN));
        hnew[m][i] = (1.f - zg) * n + zg * (float)hb[b][c];
      }
    }
    __syncthreads();
#pragma unroll
    for (int m = 0; m < 2; ++m)
#pragma unroll
      for (int i = 0; i < 8; ++i) {
        int b = m * 16 + rh + i;
        hb[b][c] = (_Float16)hnew[m][i];
        if (hs) hs[((long)b * Ts + tt) * H_ + c] = hnew[m][i];
      }
    __syncthreads();
  }
  for (int i = tid; i < B_ * H_; i += 1024) {
    int b = i >> 9, cc = i & (H_ - 1);
    hlast[(long)b * hl_ld + hl_off + cc] = (float)hb[b][cc];
  }
}

// ---------------- z / reparam / concat ----------------
__global__ void k_zcomb(const float* mu_r, const float* lv_r, const float* eps_r,
                        const float* mu_n, const float* lv_n, const float* eps_n,
                        const float* chroma,
                        float* out_std_r, float* out_std_n,
                        float* out_zr, float* out_zn,
                        float* zr_ws, float* zn_ws, float* zcat) {
  int i = blockIdx.x * blockDim.x + threadIdx.x;
  if (i < B_ * Z_) {
    int b = i / Z_, k = i % Z_;
    float sr = __expf(lv_r[i]), sn = __expf(lv_n[i]);
    float zr = mu_r[i] + sr * eps_r[i];
    float zn = mu_n[i] + sn * eps_n[i];
    out_std_r[i] = sr; out_std_n[i] = sn;
    out_zr[i] = zr;    out_zn[i] = zn;
    zr_ws[i] = zr;     zn_ws[i] = zn;
    zcat[b * ZC + k] = zr;
    zcat[b * ZC + Z_ + k] = zn;
  }
  if (i < B_ * CHR_) {
    int b = i / CHR_, k = i % CHR_;
    zcat[b * ZC + 2 * Z_ + k] = chroma[i];
  }
}

// ---------------- log_softmax over TIME axis (axis=1 of [B,T,C]) ----------------
__global__ void k_lsm_time(const float* __restrict__ logits, int C, float* __restrict__ out) {
  int b = blockIdx.x / C, c = blockIdx.x % C;
  int t = threadIdx.x;
  __shared__ float red[T_];
  float v = logits[((long)b * T_ + t) * C + c];
  red[t] = v; __syncthreads();
  for (int s = T_ / 2; s > 0; s >>= 1) { if (t < s) red[t] = fmaxf(red[t], red[t + s]); __syncthreads(); }
  float mx = red[0]; __syncthreads();
  red[t] = __expf(v - mx); __syncthreads();
  for (int s = T_ / 2; s > 0; s >>= 1) { if (t < s) red[t] += red[t + s]; __syncthreads(); }
  float lse = mx + __logf(red[0]);
  out[((long)b * T_ + t) * C + c] = v - lse;
}

// ---------------- global autoregressive decoder (persistent, 32 waves) ----------------
// LDS hh: [:,0:512]=h1, [:,512:1024]=h2.  cell_g2 fuses K=1024 over [h1|h2] for r/z gates.
__global__ void __launch_bounds__(1024)
k_dec_g(const float* __restrict__ zpart,     // [32,1536] = z@Wih_g[:,342:]^T + bih
        const float* __restrict__ Wih_g,     // [1536,622] f32 (one-hot column gather)
        const _Float16* __restrict__ Whh1,   // [1536,512] f16
        const float* __restrict__ bhh1,
        const _Float16* __restrict__ W2,     // [1536,1024] f16 = [Wih2 | Whh2]
        const float* __restrict__ bih2, const float* __restrict__ bhh2,
        const float* __restrict__ Wout,      // [342,512] f32
        const float* __restrict__ bout,
        const float* __restrict__ t0,        // [32,512]
        float* __restrict__ lbuf,            // [32,342] scratch
        float* __restrict__ gout) {          // [B,T,ROLL]
  __shared__ alignas(32) _Float16 hh[B_][1024];
  __shared__ int sidx[B_];
  int tid = threadIdx.x;
  for (int i = tid; i < B_ * H_; i += 1024) { int b = i >> 9, c = i & (H_ - 1); hh[b][c] = (_Float16)t0[i]; }
  if (tid < B_) sidx[tid] = ROLL_ - 1;           // out0 one-hot at last class
  __syncthreads();
  int w = tid >> 5, lane = tid & 31;
  int colL = lane & 15, rh = (lane >> 4) * 8;
  int c = w * 16 + colL;
  float b1R = bhh1[c], b1Z = bhh1[H_ + c], b1N = bhh1[2 * H_ + c];
  float b2R = bih2[c] + bhh2[c], b2Z = bih2[H_ + c] + bhh2[H_ + c];
  float b2Ni = bih2[2 * H_ + c], b2Nh = bhh2[2 * H_ + c];
  const _Float16* BR = Whh1 + (long)(w * 16) * H_;
  const _Float16* BZ = Whh1 + (long)(H_ + w * 16) * H_;
  const _Float16* BN = Whh1 + (long)(2 * H_ + w * 16) * H_;
  const _Float16* CR = W2 + (long)(w * 16) * 1024;
  const _Float16* CZ = W2 + (long)(H_ + w * 16) * 1024;
  const _Float16* CN = W2 + (long)(2 * H_ + w * 16) * 1024;
  __builtin_prefetch(BR, 0, 1);
  __builtin_prefetch(CR, 0, 1);
  for (int t = 0; t < T_; ++t) {
    // ---- cell_g: gh1 = h1 @ Whh1^T ----
    v8f aR0 = {}, aZ0 = {}, aN0 = {}, aR1 = {}, aZ1 = {}, aN1 = {};
    for (int k = 0; k < H_; k += 32) {
      v16h a0 = ldfragA(&hh[0][0] + k, 1024);
      v16h a1 = ldfragA(&hh[16][0] + k, 1024);
      v16h fR = ldfragB(BR + k, H_), fZ = ldfragB(BZ + k, H_), fN = ldfragB(BN + k, H_);
      aR0 = wmma32(a0, fR, aR0); aR1 = wmma32(a1, fR, aR1);
      aZ0 = wmma32(a0, fZ, aZ0); aZ1 = wmma32(a1, fZ, aZ1);
      aN0 = wmma32(a0, fN, aN0); aN1 = wmma32(a1, fN, aN1);
    }
    float h1new[2][8];
#pragma unroll
    for (int m = 0; m < 2; ++m) {
      v8f aR = m ? aR1 : aR0, aZ = m ? aZ1 : aZ0, aN = m ? aN1 : aN0;
#pragma unroll
      for (int i = 0; i < 8; ++i) {
        int b = m * 16 + rh + i;
        int oh = sidx[b];
        float gR = zpart[(long)b * G3H + c]           + Wih_g[(long)c * GIN + oh];
        float gZ = zpart[(long)b * G3H + H_ + c]      + Wih_g[(long)(H_ + c) * GIN + oh];
        float gN = zpart[(long)b * G3H + 2 * H_ + c]  + Wih_g[(long)(2 * H_ + c) * GIN + oh];
        float r  = sigmf(gR + aR[i] + b1R);
        float zg = sigmf(gZ + aZ[i] + b1Z);
        float n  = tanhf(gN + r * (aN[i] + b1N));
        h1new[m][i] = (1.f - zg) * n + zg * (float)hh[b][c];
      }
    }
    __syncthreads();
#pragma unroll
    for (int m = 0; m < 2; ++m)
#pragma unroll
      for (int i = 0; i < 8; ++i) {
        int b = m * 16 + rh + i;
        hh[b][c] = (_Float16)h1new[m][i];
        if (t == 0) hh[b][H_ + c] = (_Float16)h1new[m][i];   // h2 init = h1 at step 0
      }
    __syncthreads();
    // ---- cell_g2: r/z fused over K=1024; n-gate split (inn: k<512 on h1, hn: k>=512 on h2) ----
    v8f cR0 = {}, cZ0 = {}, cR1 = {}, cZ1 = {}, cI0 = {}, cI1 = {}, cH0 = {}, cH1 = {};
    for (int k = 0; k < 1024; k += 32) {
      v16h a0 = ldfragA(&hh[0][0] + k, 1024);
      v16h a1 = ldfragA(&hh[16][0] + k, 1024);
      v16h fR = ldfragB(CR + k, 1024), fZ = ldfragB(CZ + k, 1024), fN = ldfragB(CN + k, 1024);
      cR0 = wmma32(a0, fR, cR0); cR1 = wmma32(a1, fR, cR1);
      cZ0 = wmma32(a0, fZ, cZ0); cZ1 = wmma32(a1, fZ, cZ1);
      if (k < H_) { cI0 = wmma32(a0, fN, cI0); cI1 = wmma32(a1, fN, cI1); }
      else        { cH0 = wmma32(a0, fN, cH0); cH1 = wmma32(a1, fN, cH1); }
    }
    float h2new[2][8];
#pragma unroll
    for (int m = 0; m < 2; ++m) {
      v8f cR = m ? cR1 : cR0, cZ = m ? cZ1 : cZ0, cI = m ? cI1 : cI0, cH = m ? cH1 : cH0;
#pragma unroll
      for (int i = 0; i < 8; ++i) {
        int b = m * 16 + rh + i;
        float r  = sigmf(cR[i] + b2R);
        float zg = sigmf(cZ[i] + b2Z);
        float n  = tanhf(cI[i] + b2Ni + r * (cH[i] + b2Nh));
        h2new[m][i] = (1.f - zg) * n + zg * (float)hh[b][H_ + c];
      }
    }
    __syncthreads();
#pragma unroll
    for (int m = 0; m < 2; ++m)
#pragma unroll
      for (int i = 0; i < 8; ++i) { int b = m * 16 + rh + i; hh[b][H_ + c] = (_Float16)h2new[m][i]; }
    __syncthreads();
    // ---- logits = h2 @ Wout^T + bout ----
    for (int idx = tid; idx < B_ * ROLL_; idx += 1024) {
      int b = idx / ROLL_, n = idx % ROLL_;
      const float* wr = Wout + (long)n * H_;
      float s = bout[n];
      for (int k = 0; k < H_; ++k) s += (float)hh[b][H_ + k] * wr[k];
      lbuf[idx] = s;
    }
    __syncthreads();
    // ---- per-batch log_softmax over classes + argmax (wave b handles batch b) ----
    {
      int b = w;
      float mx = -3.4e38f; int mi = 0;
      for (int n = lane; n < ROLL_; n += 32) {
        float v = lbuf[b * ROLL_ + n];
        if (v > mx) { mx = v; mi = n; }
      }
      for (int off = 16; off > 0; off >>= 1) {
        float omx = __shfl_xor(mx, off, 32);
        int   omi = __shfl_xor(mi, off, 32);
        if (omx > mx || (omx == mx && omi < mi)) { mx = omx; mi = omi; }
      }
      float sum = 0.f;
      for (int n = lane; n < ROLL_; n += 32) sum += __expf(lbuf[b * ROLL_ + n] - mx);
      for (int off = 16; off > 0; off >>= 1) sum += __shfl_xor(sum, off, 32);
      float lse = mx + __logf(sum);
      for (int n = lane; n < ROLL_; n += 32)
        gout[((long)b * T_ + t) * ROLL_ + n] = lbuf[b * ROLL_ + n] - lse;
      if (lane == 0) sidx[b] = mi;
    }
    __syncthreads();
  }
}

// ---------------- host orchestration ----------------
enum {
  IN_X = 0, IN_RHY, IN_NOTE, IN_CHR, IN_EPSR, IN_EPSN,
  P_RF_WIH, P_RF_WHH, P_RF_BIH, P_RF_BHH,
  P_RB_WIH, P_RB_WHH, P_RB_BIH, P_RB_BHH,
  P_NF_WIH, P_NF_WHH, P_NF_BIH, P_NF_BHH,
  P_NB_WIH, P_NB_WHH, P_NB_BIH, P_NB_BHH,
  P_MUR_W, P_MUR_B, P_VARR_W, P_VARR_B,
  P_MUN_W, P_MUN_B, P_VARN_W, P_VARN_B,
  P_DR_WIH, P_DR_WHH, P_DR_BIH, P_DR_BHH,
  P_DN_WIH, P_DN_WHH, P_DN_BIH, P_DN_BHH,
  P_INITR_W, P_INITR_B, P_INITN_W, P_INITN_B,
  P_OUTR_W, P_OUTR_B, P_OUTN_W, P_OUTN_B,
  P_INITG_W, P_INITG_B,
  P_G_WIH, P_G_WHH, P_G_BIH, P_G_BHH,
  P_G2_WIH, P_G2_WHH, P_G2_BIH, P_G2_BHH,
  P_OUTG_W, P_OUTG_B
};

extern "C" void kernel_launch(void* const* d_in, const int* in_sizes, int n_in,
                              void* d_out, int out_size, void* d_ws, size_t ws_size,
                              hipStream_t stream) {
  (void)in_sizes; (void)n_in; (void)out_size; (void)ws_size;
  const float* F[64];
  for (int i = 0; i < 58 && i < n_in; ++i) F[i] = (const float*)d_in[i];
  float* out = (float*)d_out;

  // d_out regions (return order: g_out, r_out, n_out, mu_r, std_r, mu_n, std_n, z_r, z_n)
  float* o_g    = out;
  float* o_r    = out + (long)B_ * T_ * ROLL_;
  float* o_n    = o_r + (long)B_ * T_ * RHY_;
  float* o_mur  = o_n + (long)B_ * T_ * NOTE_;
  float* o_stdr = o_mur + B_ * Z_;
  float* o_mun  = o_stdr + B_ * Z_;
  float* o_stdn = o_mun + B_ * Z_;
  float* o_zr   = o_stdn + B_ * Z_;
  float* o_zn   = o_zr + B_ * Z_;

  // bump allocator over d_ws
  char* wp = (char*)d_ws;
  auto alloc = [&](size_t bytes) -> void* {
    void* p = (void*)wp; wp += (bytes + 255) & ~(size_t)255; return p;
  };
  const long MT = (long)B_ * T_;                         // 8192
  float*     gi      = (float*)alloc(MT * G3H * 4);      // reused by all GRUs
  float*     hs      = (float*)alloc(MT * H_ * 4);       // sub-decoder hidden seq
  float*     lsub    = (float*)alloc(MT * NOTE_ * 4);    // sub-decoder logits
  _Float16*  xh      = (_Float16*)alloc(MT * KX * 2);
  _Float16*  wihbuf  = (_Float16*)alloc((long)G3H * KX * 2);   // reused per GRU
  _Float16*  whhbuf  = (_Float16*)alloc((long)G3H * H_ * 2);   // reused per GRU
  _Float16*  din     = (_Float16*)alloc(MT * KD * 2);
  _Float16*  w2      = (_Float16*)alloc((long)G3H * 1024 * 2);
  float*     xr_cat  = (float*)alloc(B_ * 2 * H_ * 4);
  float*     xn_cat  = (float*)alloc(B_ * 2 * H_ * 4);
  float*     lv_r    = (float*)alloc(B_ * Z_ * 4);
  float*     lv_n    = (float*)alloc(B_ * Z_ * 4);
  float*     zr_ws   = (float*)alloc(B_ * Z_ * 4);
  float*     zn_ws   = (float*)alloc(B_ * Z_ * 4);
  float*     zcat    = (float*)alloc(B_ * ZC * 4);
  float*     h0_r    = (float*)alloc(B_ * H_ * 4);
  float*     h0_n    = (float*)alloc(B_ * H_ * 4);
  float*     t0      = (float*)alloc(B_ * H_ * 4);
  float*     zpart   = (float*)alloc(B_ * G3H * 4);
  float*     lbuf    = (float*)alloc(B_ * ROLL_ * 4);
  float*     hscr    = (float*)alloc(B_ * H_ * 4);       // throwaway hlast

  auto zeroh = [&](_Float16* p, long n) {
    k_zero_h<<<(unsigned)((n + 255) / 256), 256, 0, stream>>>(p, n);
  };
  auto pack = [&](const float* s, int sld, int soff, _Float16* d, int dld, int doff, int r, int c) {
    long n = (long)r * c;
    k_pack<<<(unsigned)((n + 255) / 256), 256, 0, stream>>>(s, sld, soff, d, dld, doff, r, c);
  };
  auto gemm = [&](const _Float16* A, int lda, const _Float16* Bw, int ldb,
                  const float* bias, float* C, int ldc, int M, int N, int Kp) {
    int tiles = ((M + 15) / 16) * (N / 16);
    k_gemm_f16<<<(tiles + 7) / 8, 256, 0, stream>>>(A, lda, Bw, ldb, bias, C, ldc, M, N, Kp);
  };
  auto lin = [&](const float* A, int lda, const float* W, int ldw, int woff,
                 const float* bias, float* C, int ldc, int M, int N, int K) {
    long n = (long)M * N;
    k_linear<<<(unsigned)((n + 255) / 256), 256, 0, stream>>>(A, lda, W, ldw, woff, bias, C, ldc, M, N, K);
  };

  // ---- stage x as padded f16 ----
  zeroh(xh, MT * KX);
  pack(F[IN_X], ROLL_, 0, xh, KX, 0, (int)MT, ROLL_);

  // ---- 4 bidirectional-encoder GRUs ----
  struct Enc { int wih, whh, bih, bhh, rev; float* cat; int off; };
  Enc encs[4] = {
    { P_RF_WIH, P_RF_WHH, P_RF_BIH, P_RF_BHH, 0, xr_cat, 0   },
    { P_RB_WIH, P_RB_WHH, P_RB_BIH, P_RB_BHH, 1, xr_cat, H_  },
    { P_NF_WIH, P_NF_WHH, P_NF_BIH, P_NF_BHH, 0, xn_cat, 0   },
    { P_NB_WIH, P_NB_WHH, P_NB_BIH, P_NB_BHH, 1, xn_cat, H_  },
  };
  for (int e = 0; e < 4; ++e) {
    zeroh(wihbuf, (long)G3H * KX);
    pack(F[encs[e].wih], ROLL_, 0, wihbuf, KX, 0, G3H, ROLL_);
    pack(F[encs[e].whh], H_, 0, whhbuf, H_, 0, G3H, H_);
    gemm(xh, KX, wihbuf, KX, F[encs[e].bih], gi, G3H, (int)MT, G3H, KX);
    k_gru_seq<<<1, 1024, 0, stream>>>(gi, whhbuf, F[encs[e].bhh], nullptr, nullptr,
                                      encs[e].cat, 2 * H_, encs[e].off, T_, encs[e].rev);
  }

  // ---- mu/var heads ----
  lin(xr_cat, 2 * H_, F[P_MUR_W], 2 * H_, 0, F[P_MUR_B], o_mur, Z_, B_, Z_, 2 * H_);
  lin(xr_cat, 2 * H_, F[P_VARR_W], 2 * H_, 0, F[P_VARR_B], lv_r, Z_, B_, Z_, 2 * H_);
  lin(xn_cat, 2 * H_, F[P_MUN_W], 2 * H_, 0, F[P_MUN_B], o_mun, Z_, B_, Z_, 2 * H_);
  lin(xn_cat, 2 * H_, F[P_VARN_W], 2 * H_, 0, F[P_VARN_B], lv_n, Z_, B_, Z_, 2 * H_);
  k_zcomb<<<(B_ * Z_ + 255) / 256, 256, 0, stream>>>(
      o_mur, lv_r, F[IN_EPSR], o_mun, lv_n, F[IN_EPSN], F[IN_CHR],
      o_stdr, o_stdn, o_zr, o_zn, zr_ws, zn_ws, zcat);

  // ---- decoder initial states ----
  lin(zr_ws, Z_, F[P_INITR_W], Z_, 0, F[P_INITR_B], h0_r, H_, B_, H_, Z_);
  lin(zn_ws, Z_, F[P_INITN_W], Z_, 0, F[P_INITN_B], h0_n, H_, B_, H_, Z_);
  lin(zcat, ZC, F[P_INITG_W], ZC, 0, F[P_INITG_B], t0, H_, B_, H_, ZC);
  // zpart = z @ Wih_g[:,342:]^T + bih_g   (one-hot part gathered per step)
  lin(zcat, ZC, F[P_G_WIH], GIN, ROLL_, F[P_G_BIH], zpart, G3H, B_, G3H, ZC);

  // ---- sub-decoder (rhythm) ----
  k_dec_in<<<(unsigned)((MT * KD + 255) / 256), 256, 0, stream>>>(F[IN_RHY], RHY_, zr_ws, din);
  zeroh(wihbuf, (long)G3H * KD);
  pack(F[P_DR_WIH], Z_ + RHY_, 0, wihbuf, KD, 0, G3H, Z_ + RHY_);
  pack(F[P_DR_WHH], H_, 0, whhbuf, H_, 0, G3H, H_);
  gemm(din, KD, wihbuf, KD, F[P_DR_BIH], gi, G3H, (int)MT, G3H, KD);
  k_gru_seq<<<1, 1024, 0, stream>>>(gi, whhbuf, F[P_DR_BHH], h0_r, hs, hscr, H_, 0, T_, 0);
  lin(hs, H_, F[P_OUTR_W], H_, 0, F[P_OUTR_B], lsub, RHY_, (int)MT, RHY_, H_);
  k_lsm_time<<<B_ * RHY_, T_, 0, stream>>>(lsub, RHY_, o_r);

  // ---- sub-decoder (note) ----
  k_dec_in<<<(unsigned)((MT * KD + 255) / 256), 256, 0, stream>>>(F[IN_NOTE], NOTE_, zn_ws, din);
  zeroh(wihbuf, (long)G3H * KD);
  pack(F[P_DN_WIH], Z_ + NOTE_, 0, wihbuf, KD, 0, G3H, Z_ + NOTE_);
  pack(F[P_DN_WHH], H_, 0, whhbuf, H_, 0, G3H, H_);
  gemm(din, KD, wihbuf, KD, F[P_DN_BIH], gi, G3H, (int)MT, G3H, KD);
  k_gru_seq<<<1, 1024, 0, stream>>>(gi, whhbuf, F[P_DN_BHH], h0_n, hs, hscr, H_, 0, T_, 0);
  lin(hs, H_, F[P_OUTN_W], H_, 0, F[P_OUTN_B], lsub, NOTE_, (int)MT, NOTE_, H_);
  k_lsm_time<<<B_ * NOTE_, T_, 0, stream>>>(lsub, NOTE_, o_n);

  // ---- global autoregressive decoder ----
  pack(F[P_G_WHH], H_, 0, whhbuf, H_, 0, G3H, H_);         // Whh of cell_g
  pack(F[P_G2_WIH], H_, 0, w2, 1024, 0, G3H, H_);          // [Wih2 | Whh2]
  pack(F[P_G2_WHH], H_, 0, w2, 1024, H_, G3H, H_);
  k_dec_g<<<1, 1024, 0, stream>>>(zpart, F[P_G_WIH], whhbuf, F[P_G_BHH],
                                  w2, F[P_G2_BIH], F[P_G2_BHH],
                                  F[P_OUTG_W], F[P_OUTG_B], t0, lbuf, o_g);
}
